// MoE_layer_megatron_10892037063035
// MI455X (gfx1250) — matrix operand order repaired and checked by reading, not compiled, not run on hardware
//
#include <hip/hip_runtime.h>
#include <hip/hip_bf16.h>

// Problem constants (from the reference): S=2048, B=4, H=1024, F=4096, E=8, K=2
#define S_ 2048
#define B_ 4
#define T_ (S_ * B_)   // 8192 tokens
#define H_ 1024
#define F_ 4096
#define E_ 8

typedef __attribute__((ext_vector_type(16))) __bf16 v16bf;
typedef __attribute__((ext_vector_type(8)))  __bf16 v8bf;
typedef __attribute__((ext_vector_type(8)))  float  v8f;
typedef __attribute__((ext_vector_type(4)))  unsigned int u32x4;
typedef __attribute__((ext_vector_type(8)))  int i32x8;
typedef __attribute__((ext_vector_type(4)))  int i32x4;

#if defined(__has_builtin)
#if __has_builtin(__builtin_amdgcn_tensor_load_to_lds) && \
    __has_builtin(__builtin_amdgcn_s_wait_tensorcnt)
#define USE_TDM 1
#endif
#endif

__device__ __forceinline__ v16bf frag_cat(v8bf lo, v8bf hi) {
  return __builtin_shufflevector(lo, hi, 0, 1, 2, 3, 4, 5, 6, 7,
                                         8, 9, 10, 11, 12, 13, 14, 15);
}

__device__ __forceinline__ v8f zero8() {
  v8f z;
#pragma unroll
  for (int i = 0; i < 8; i++) z[i] = 0.0f;
  return z;
}

__device__ __forceinline__ float gelu_tanh(float x) {
  // jax.nn.gelu(approximate=True)
  float x3 = x * x * x;
  return 0.5f * x * (1.0f + tanhf(0.7978845608028654f * (x + 0.044715f * x3)));
}

// ---------------------------------------------------------------------------
// Router: logits = x @ router_w, softmax, top-2 -> dense gate matrix [T, E]
// ---------------------------------------------------------------------------
__global__ void router_kernel(const float* __restrict__ x,
                              const float* __restrict__ rw,
                              float* __restrict__ gates) {
  const int wid = threadIdx.x >> 5;
  const int lane = threadIdx.x & 31;
  const int t = blockIdx.x * 8 + wid;

  float acc[E_];
#pragma unroll
  for (int e = 0; e < E_; e++) acc[e] = 0.0f;

  for (int h = lane; h < H_; h += 32) {
    float xv = x[(size_t)t * H_ + h];
#pragma unroll
    for (int e = 0; e < E_; e++) acc[e] += xv * rw[h * E_ + e];
  }
#pragma unroll
  for (int off = 16; off > 0; off >>= 1) {
#pragma unroll
    for (int e = 0; e < E_; e++) acc[e] += __shfl_xor(acc[e], off, 32);
  }

  float mx = acc[0];
#pragma unroll
  for (int e = 1; e < E_; e++) mx = fmaxf(mx, acc[e]);
  float p[E_];
  float s = 0.0f;
#pragma unroll
  for (int e = 0; e < E_; e++) { p[e] = __expf(acc[e] - mx); s += p[e]; }
  float inv = 1.0f / s;
#pragma unroll
  for (int e = 0; e < E_; e++) p[e] *= inv;

  int i1 = 0;
#pragma unroll
  for (int e = 1; e < E_; e++) if (p[e] > p[i1]) i1 = e;
  int i2 = (i1 == 0) ? 1 : 0;
#pragma unroll
  for (int e = 0; e < E_; e++) if (e != i1 && p[e] > p[i2]) i2 = e;

  if (lane == 0) {
#pragma unroll
    for (int e = 0; e < E_; e++)
      gates[(size_t)t * E_ + e] = (e == i1 || e == i2) ? p[e] : 0.0f;
  }
}

// ---------------------------------------------------------------------------
// Elementwise fp32 -> bf16 for activations
// ---------------------------------------------------------------------------
__global__ void convert_x_kernel(const float* __restrict__ x,
                                 __bf16* __restrict__ xb) {
  size_t i = ((size_t)blockIdx.x * blockDim.x + threadIdx.x) * 8;
#pragma unroll
  for (int j = 0; j < 8; j++) xb[i + j] = (__bf16)x[i + j];
}

// ---------------------------------------------------------------------------
// Tiled transpose + fp32 -> bf16:  out[z][c][r] = (bf16) in[z][r][c]
// ---------------------------------------------------------------------------
__global__ void transpose_bf16_kernel(const float* __restrict__ in,
                                      __bf16* __restrict__ out,
                                      int R, int C) {
  __shared__ float tile[32][33];
  const float* inm = in + (size_t)blockIdx.z * R * C;
  __bf16* outm = out + (size_t)blockIdx.z * R * C;

  int c = blockIdx.x * 32 + threadIdx.x;
#pragma unroll
  for (int j = threadIdx.y; j < 32; j += 8) {
    int r = blockIdx.y * 32 + j;
    tile[j][threadIdx.x] = inm[(size_t)r * C + c];
  }
  __syncthreads();
  int ro = blockIdx.y * 32 + threadIdx.x;
#pragma unroll
  for (int j = threadIdx.y; j < 32; j += 8) {
    int co = blockIdx.x * 32 + j;
    outm[(size_t)co * R + ro] = (__bf16)tile[threadIdx.x][j];
  }
}

// ---------------------------------------------------------------------------
// Fused MoE kernel.
//
// x tile staged by the Tensor Data Mover with pad_enable: 16B pad per 1024B
//   -> LDS row stride 2080B (520 dw). Element offset: r*1040 + c + 8*(c>>9).
//   Fragment runs (8 bf16 at c % 8 == 0) stay 16B aligned for ds_load_b128.
//
// WMMA fragment layouts (wave32, v_wmma_f32_16x16x32_bf16):
//  A (16x32 bf16):  lane L -> row L%16, elems k = 16*(j/8) + 8*(L/16) + j%8
//  B (32x16 bf16):  lane L -> col L%16, elems k = 16*(j/8) + 8*(L/16) + j%8
//  C/D (16x16 f32): lane L -> col L%16, VGPR v -> row v + 8*(L/16)
// ---------------------------------------------------------------------------
#define XLDS_STRIDE 1040   // elements per padded LDS row (1024 + 2x8 pad)
#define HLDS_STRIDE 136    // 128 + 8 bf16 pad

__device__ __forceinline__ const v8bf* xrun(const __bf16* x_lds, int r, int c) {
  return (const v8bf*)(x_lds + r * XLDS_STRIDE + c + 8 * (c >> 9));
}

__device__ __forceinline__ v16bf loadB16(const __bf16* p) {
  return frag_cat(*(const v8bf*)p, *(const v8bf*)(p + 16));
}

__global__ __launch_bounds__(256)
void moe_fused_kernel(const __bf16* __restrict__ xb,    // [T][H]
                      const __bf16* __restrict__ w1t,   // [E][F][H]
                      const __bf16* __restrict__ w2t,   // [E][H][F]
                      const float*  __restrict__ gates, // [T][E]
                      float* __restrict__ out) {        // [T][H]
  extern __shared__ char smem[];
  __bf16* x_lds = (__bf16*)smem;                         // 32 x XLDS_STRIDE
  __bf16* h_lds = x_lds + 32 * XLDS_STRIDE;              // 32 x HLDS_STRIDE
  float* gate_lds = (float*)(h_lds + 32 * HLDS_STRIDE);  // 32 x E

  const int tid = threadIdx.x;
  const int wv = tid >> 5;       // wave 0..7 -> output cols [wv*128, wv*128+128)
  const int lane = tid & 31;
  const int half = lane >> 4;
  const int n16 = lane & 15;
  const int t0 = blockIdx.x * 32;

#ifdef USE_TDM
  // One wave issues a single 2D TDM descriptor for the whole 32x1024 bf16 tile.
  if (wv == 0) {
    unsigned long long ga = (unsigned long long)(uintptr_t)(xb + (size_t)t0 * H_);
    u32x4 g0;
    g0[0] = 1u;                                   // count=1, user mode
    g0[1] = 0u;                                   // lds_addr (dynamic LDS base)
    g0[2] = (unsigned int)(ga & 0xffffffffull);   // global_addr[31:0]
    g0[3] = (unsigned int)((ga >> 32) & 0x01ffffffull) | (2u << 30);  // type=2
    i32x8 g1;
    g1[0] = (1 << 16)      // data_size = 2 bytes
          | (1 << 20)      // pad_enable
          | (7 << 22)      // pad_interval code 7 = every 256 dwords (1024B)
          | (3 << 25);     // pad_amount  code 3 = 4 dwords (16B)
    g1[1] = (H_ & 0xffff) << 16;                  // tensor_dim0[15:0]
    g1[2] = (H_ >> 16) | ((T_ & 0xffff) << 16);   // dim0[31:16] | dim1[15:0]
    g1[3] = (T_ >> 16) | (H_ << 16);              // dim1[31:16] | tile_dim0
    g1[4] = 32;                                   // tile_dim1=32, tile_dim2=0
    g1[5] = H_;                                   // tensor_dim0_stride lo32
    g1[6] = 0;                                    // stride0 hi | stride1 lo
    g1[7] = 0;                                    // stride1 hi
    i32x4 gz4;
    gz4[0] = 0; gz4[1] = 0; gz4[2] = 0; gz4[3] = 0;   // groups 2/3: unused dims
    i32x8 gz8;
#pragma unroll
    for (int i = 0; i < 8; i++) gz8[i] = 0;
    // clang-23 / therock lane: 6-arg form (g0, g1, g2, g3, extra, cpol)
    __builtin_amdgcn_tensor_load_to_lds(g0, g1, gz4, gz4, gz8, 0);
    __builtin_amdgcn_s_wait_tensorcnt(0);
  }
#else
  for (int i = tid; i < 32 * (H_ / 8); i += 256) {
    int r = i >> 7;
    int c = (i & 127) * 8;
    *(v8bf*)(x_lds + r * XLDS_STRIDE + c + 8 * (c >> 9)) =
        *(const v8bf*)&xb[(size_t)(t0 + r) * H_ + c];
  }
#endif
  if (tid < 32 * E_) gate_lds[tid] = gates[(size_t)t0 * E_ + tid];
  __syncthreads();

  v8f acc0[8], acc1[8];
#pragma unroll
  for (int i = 0; i < 8; i++) { acc0[i] = zero8(); acc1[i] = zero8(); }

  const __bf16* hrow0 = h_lds + n16 * HLDS_STRIDE + 8 * half;
  const __bf16* hrow1 = h_lds + (16 + n16) * HLDS_STRIDE + 8 * half;

  for (int e = 0; e < E_; e++) {
    const __bf16* w1e = w1t + (size_t)e * F_ * H_;
    const __bf16* w2e = w2t + (size_t)e * H_ * F_;

    for (int ck = 0; ck < F_ / 128; ck++) {
      const int f0 = ck * 128;

      // ---- fc1: h[32 x 16] at f = f0 + wv*16; B stream double-buffered ----
      v8f h0 = zero8(), h1 = zero8();
      const __bf16* brow = w1e + (size_t)(f0 + wv * 16 + n16) * H_ + 8 * half;
      v16bf Bc = loadB16(brow);
      v16bf Bn = loadB16(brow + 32);
      for (int k0 = 0; k0 < H_; k0 += 32) {
        const int c0 = k0 + 8 * half;
        v16bf A0 = frag_cat(*xrun(x_lds, n16, c0), *xrun(x_lds, n16, c0 + 16));
        v16bf A1 = frag_cat(*xrun(x_lds, 16 + n16, c0),
                            *xrun(x_lds, 16 + n16, c0 + 16));
        v16bf Bp = Bn;
        if (k0 + 64 < H_) Bn = loadB16(brow + k0 + 64);  // prefetch k+2
        h0 = __builtin_amdgcn_wmma_f32_16x16x32_bf16(false, A0, false, Bc,
                                                     (short)0, h0, false, false);
        h1 = __builtin_amdgcn_wmma_f32_16x16x32_bf16(false, A1, false, Bc,
                                                     (short)0, h1, false, false);
        Bc = Bp;
      }

      // gelu + fold router gate into hidden, store bf16 tile to LDS
#pragma unroll
      for (int v = 0; v < 8; v++) {
        int m = v + 8 * half;
        h_lds[m * HLDS_STRIDE + wv * 16 + n16] =
            (__bf16)(gate_lds[m * E_ + e] * gelu_tanh(h0[v]));
        h_lds[(16 + m) * HLDS_STRIDE + wv * 16 + n16] =
            (__bf16)(gate_lds[(16 + m) * E_ + e] * gelu_tanh(h1[v]));
      }
      __syncthreads();

      // ---- fc2: acc[32 x 128] += h[32 x 128] @ w2[f0:f0+128, wv*128:+128] ----
      // Batch all 8 B fragments per k-step so one wait covers the clause and
      // the previous group's 16 WMMAs overlap these loads.
      for (int k0 = 0; k0 < 128; k0 += 32) {
        v16bf A0 = frag_cat(*(const v8bf*)(hrow0 + k0),
                            *(const v8bf*)(hrow0 + k0 + 16));
        v16bf A1 = frag_cat(*(const v8bf*)(hrow1 + k0),
                            *(const v8bf*)(hrow1 + k0 + 16));
        v16bf Bo[8];
#pragma unroll
        for (int nt = 0; nt < 8; nt++) {
          const __bf16* b2 =
              w2e + (size_t)(wv * 128 + nt * 16 + n16) * F_ + f0 + k0 + 8 * half;
          Bo[nt] = loadB16(b2);
        }
#pragma unroll
        for (int nt = 0; nt < 8; nt++) {
          acc0[nt] = __builtin_amdgcn_wmma_f32_16x16x32_bf16(
              false, A0, false, Bo[nt], (short)0, acc0[nt], false, false);
          acc1[nt] = __builtin_amdgcn_wmma_f32_16x16x32_bf16(
              false, A1, false, Bo[nt], (short)0, acc1[nt], false, false);
        }
      }
      __syncthreads();  // h_lds overwritten next chunk
    }
  }

  // Write the finished 32 x 1024 fp32 tile once; NT stores (no re-read).
#pragma unroll
  for (int nt = 0; nt < 8; nt++) {
#pragma unroll
    for (int v = 0; v < 8; v++) {
      int m = v + 8 * half;
      int col = wv * 128 + nt * 16 + n16;
      __builtin_nontemporal_store(acc0[nt][v], &out[(size_t)(t0 + m) * H_ + col]);
      __builtin_nontemporal_store(acc1[nt][v],
                                  &out[(size_t)(t0 + 16 + m) * H_ + col]);
    }
  }
}

// ---------------------------------------------------------------------------
// Host launcher
// ---------------------------------------------------------------------------
extern "C" void kernel_launch(void* const* d_in, const int* in_sizes, int n_in,
                              void* d_out, int out_size, void* d_ws,
                              size_t ws_size, hipStream_t stream) {
  (void)in_sizes; (void)n_in; (void)out_size; (void)ws_size;

  const float* x  = (const float*)d_in[0];   // [S,B,H] fp32
  const float* rw = (const float*)d_in[1];   // [H,E]
  const float* w1 = (const float*)d_in[2];   // [E,H,F]
  const float* w2 = (const float*)d_in[3];   // [E,F,H]
  float* out = (float*)d_out;                // [S,B,H] fp32

  char* ws = (char*)d_ws;
  float*  gates = (float*)ws;                                   // 256 KB
  __bf16* xb    = (__bf16*)(ws + 262144);                       // 16 MB
  __bf16* w1t   = (__bf16*)(ws + 262144 + 16777216);            // 64 MB
  __bf16* w2t   = (__bf16*)(ws + 262144 + 16777216 + 67108864); // 64 MB

  router_kernel<<<T_ / 8, 256, 0, stream>>>(x, rw, gates);
  convert_x_kernel<<<(T_ * H_) / (256 * 8), 256, 0, stream>>>(x, xb);
  transpose_bf16_kernel<<<dim3(F_ / 32, H_ / 32, E_), dim3(32, 8), 0, stream>>>(
      w1, w1t, H_, F_);
  transpose_bf16_kernel<<<dim3(H_ / 32, F_ / 32, E_), dim3(32, 8), 0, stream>>>(
      w2, w2t, F_, H_);

  size_t smem = (size_t)(32 * XLDS_STRIDE + 32 * HLDS_STRIDE) * sizeof(__bf16) +
                (size_t)(32 * E_) * sizeof(float);
  moe_fused_kernel<<<T_ / 32, 256, smem, stream>>>(xb, w1t, w2t, gates, out);
}
